// AttnBlock3D_90924457656992
// MI455X (gfx1250) — matrix operand order repaired and checked by reading, not compiled, not run on hardware
//
#include <hip/hip_runtime.h>

typedef __attribute__((ext_vector_type(16))) _Float16 v16h;
typedef __attribute__((ext_vector_type(8)))  float    v8f;

constexpr int kB   = 2;
constexpr int kC   = 64;
constexpr int kN   = 4096;          // 16*16*16 spatial positions
constexpr int kNG  = 16;            // groupnorm groups
constexpr int kCPG = kC / kNG;      // channels per group
constexpr int LDSP = 72;            // padded LDS row stride (halves): 144B = 9*16B, conflict free

#if defined(__has_builtin)
#if __has_builtin(__builtin_amdgcn_global_load_async_to_lds_b128) && \
    __has_builtin(__builtin_amdgcn_s_wait_asynccnt)
#define HAVE_ASYNC_LDS 1
#endif
#endif
#ifndef HAVE_ASYNC_LDS
#define HAVE_ASYNC_LDS 0
#endif

#if HAVE_ASYNC_LDS
// Builtin signature (from hipcc diagnostic): param0 = v4i in AS1 (global),
// param1 = v4i in AS3 (LDS), then imm offset + imm cpol.
typedef int v4i __attribute__((vector_size(16)));
typedef __attribute__((address_space(1))) v4i gv4i;
typedef __attribute__((address_space(3))) v4i lv4i;
#endif

union Frag16 { v16h h; unsigned u[8]; };

// A fragment (16x32 f16): lanes 0-15 hold M=lane, K in {0..7,16..23}; lanes 16-31 hold
// M=lane-16, K in {8..15,24..31}. Each VGPR = adjacent K pair -> packed b32 LDS loads.
__device__ inline v16h load_a_frag(const _Float16* row, int lane) {
  Frag16 f;
  const int lo = (lane & 16) ? 8 : 0;
#pragma unroll
  for (int v = 0; v < 8; ++v) {
    const int k = ((v & 4) ? 16 : 0) + lo + (v & 3) * 2;
    f.u[v] = *(const unsigned*)(row + k);
  }
  return f.h;
}

// B fragment (32x16 f16): lane = column N, lanes 0-15 K in {0..7,16..23}, lanes 16-31 K in {8..15,24..31},
// K contiguous -> row points at [n][k0], packed b32 loads.
__device__ inline v16h load_b_frag(const _Float16* row, int lane) {
  Frag16 f;
  const int base = (lane & 16) ? 8 : 0;
#pragma unroll
  for (int v = 0; v < 8; ++v)
    f.u[v] = *(const unsigned*)(row + base + ((v & 4) ? 16 : 0) + (v & 3) * 2);
  return f.h;
}

__device__ inline v8f wmma_f16(v16h a, v16h b, v8f c) {
  return __builtin_amdgcn_wmma_f32_16x16x32_f16(false, a, false, b, (short)0, c,
                                                false, false);
}

// ---------------------------------------------------------------------------
// Kernel 1: GroupNorm statistics. One block per (batch, group).
// ---------------------------------------------------------------------------
__global__ __launch_bounds__(256) void gn_stats_kernel(const float* __restrict__ x,
                                                       float* __restrict__ stats) {
  const int bg = blockIdx.x;                 // 0..kB*kNG-1
  const int b  = bg / kNG;
  const int g  = bg % kNG;
  const float* xp = x + ((size_t)b * kC + (size_t)g * kCPG) * kN;
  float s = 0.f, ss = 0.f;
  for (int i = threadIdx.x; i < kCPG * kN; i += 256) {
    const float v = xp[i];
    s += v; ss += v * v;
  }
  __shared__ float red0[256], red1[256];
  red0[threadIdx.x] = s; red1[threadIdx.x] = ss;
  __syncthreads();
  for (int off = 128; off > 0; off >>= 1) {
    if (threadIdx.x < off) {
      red0[threadIdx.x] += red0[threadIdx.x + off];
      red1[threadIdx.x] += red1[threadIdx.x + off];
    }
    __syncthreads();
  }
  if (threadIdx.x == 0) {
    const float inv  = 1.f / (float)(kCPG * kN);
    const float mean = red0[0] * inv;
    const float var  = red1[0] * inv - mean * mean;
    stats[bg * 2 + 0] = mean;
    stats[bg * 2 + 1] = rsqrtf(var + 1e-5f);
  }
}

// ---------------------------------------------------------------------------
// Kernel 2: fused GroupNorm-apply + Q/K/V 1x1 conv (64x64 GEMM) via WMMA.
// Block = 128 threads (4 waves), 64 positions. q,k -> [b,n,c], v -> [b,c,n].
// ---------------------------------------------------------------------------
__global__ __launch_bounds__(128) void qkv_kernel(
    const float* __restrict__ x, const float* __restrict__ stats,
    const float* __restrict__ norm_w, const float* __restrict__ norm_b,
    const float* __restrict__ q_w, const float* __restrict__ q_b,
    const float* __restrict__ k_w, const float* __restrict__ k_b,
    const float* __restrict__ v_w, const float* __restrict__ v_b,
    _Float16* __restrict__ qT, _Float16* __restrict__ kTg, _Float16* __restrict__ vG) {
  const int blk = blockIdx.x;            // kB * (kN/64)
  const int b   = blk >> 6;
  const int p0  = (blk & 63) * 64;
  __shared__ _Float16 hn[64][LDSP];      // [pos][chan], normalized activations
  __shared__ _Float16 w3[3][64][LDSP];   // q/k/v weights [out][in] in f16
  const int tid = threadIdx.x;

  for (int i = tid; i < 64 * 64; i += 128) {
    const int p = i >> 6, c = i & 63;
    const float mean = stats[(b * kNG + (c >> 2)) * 2 + 0];
    const float rstd = stats[(b * kNG + (c >> 2)) * 2 + 1];
    const float xv   = x[((size_t)b * kC + c) * kN + p0 + p];
    hn[p][c] = (_Float16)(((xv - mean) * rstd) * norm_w[c] + norm_b[c]);
  }
  for (int i = tid; i < 64 * 64; i += 128) {
    const int o = i >> 6, c = i & 63;
    w3[0][o][c] = (_Float16)q_w[i];
    w3[1][o][c] = (_Float16)k_w[i];
    w3[2][o][c] = (_Float16)v_w[i];
  }
  __syncthreads();

  const int lane = tid & 31;
  const int wave = tid >> 5;             // M tile (16 positions)
  const int mrow = wave * 16 + (lane & 15);
  const float* biases[3] = {q_b, k_b, v_b};

#pragma unroll
  for (int out = 0; out < 3; ++out) {
#pragma unroll
    for (int nt = 0; nt < 4; ++nt) {
      v8f acc = {};
#pragma unroll
      for (int ks = 0; ks < 2; ++ks) {
        const v16h a  = load_a_frag(&hn[mrow][ks * 32], lane);
        const v16h bf = load_b_frag(&w3[out][nt * 16 + (lane & 15)][ks * 32], lane);
        acc = wmma_f16(a, bf, acc);
      }
      const int ncol   = nt * 16 + (lane & 15);
      const float bias = biases[out][ncol];
#pragma unroll
      for (int r = 0; r < 8; ++r) {
        const int m   = r + ((lane & 16) ? 8 : 0);
        const int pos = p0 + wave * 16 + m;
        const float d = acc[r] + bias;
        if (out == 0)      qT[((size_t)b * kN + pos) * kC + ncol] = (_Float16)d;
        else if (out == 1) kTg[((size_t)b * kN + pos) * kC + ncol] = (_Float16)d;
        else               vG[((size_t)b * kC + ncol) * kN + pos] = (_Float16)d;
      }
    }
  }
}

// ---------------------------------------------------------------------------
// Kernel 3: flash-style attention over n=4096, head dim 64.
// Block = 4 waves, 64 query rows (16 per wave). 64-wide key chunks staged into
// double-buffered LDS via async global->LDS b128 copies (ASYNCcnt), online softmax.
// ---------------------------------------------------------------------------
__global__ __launch_bounds__(128) void attn_kernel(
    const _Float16* __restrict__ qT, const _Float16* __restrict__ kTg,
    const _Float16* __restrict__ vG, _Float16* __restrict__ attnT) {
  const int blk = blockIdx.x;
  const int b   = blk >> 6;
  const int q0  = (blk & 63) * 64;
  __shared__ alignas(16) _Float16 qs[64][LDSP];         // [qpos][chan]
  __shared__ alignas(16) _Float16 kTb[2][64][LDSP];     // [kpos][chan]   (B of Q K^T)
  __shared__ alignas(16) _Float16 vsb[2][64][LDSP];     // [chan][kpos]   (B of P V^T)
  __shared__ alignas(16) _Float16 ps[4][16][LDSP];      // per-wave P tile [m][kpos]
  const int tid = threadIdx.x, lane = tid & 31, wave = tid >> 5;

  for (int i = tid; i < 64 * 64; i += 128) {
    const int p = i >> 6, c = i & 63;
    qs[p][c] = qT[((size_t)b * kN + q0 + p) * kC + c];
  }

  // stage one 64-key chunk (kT rows + v rows) into LDS buffer `buf`
  auto stage_chunk = [&](int buf, int k0) {
#if HAVE_ASYNC_LDS
    const int colh = (lane & 7) * 8;                 // 8-half (16B) group in a row
#pragma unroll
    for (int i = 0; i < 4; ++i) {
      const int row = wave * 16 + i * 4 + (lane >> 3);
      __builtin_amdgcn_global_load_async_to_lds_b128(
          (gv4i*)(kTg + ((size_t)b * kN + k0 + row) * kC + colh),
          (lv4i*)(&kTb[buf][row][colh]), 0, 0);
      __builtin_amdgcn_global_load_async_to_lds_b128(
          (gv4i*)(vG + ((size_t)b * kC + row) * kN + k0 + colh),
          (lv4i*)(&vsb[buf][row][colh]), 0, 0);
    }
#else
    for (int i = tid; i < 64 * 64; i += 128) {
      const int r = i >> 6, p = i & 63;
      kTb[buf][r][p] = kTg[((size_t)b * kN + k0 + r) * kC + p];
      vsb[buf][r][p] = vG[((size_t)b * kC + r) * kN + k0 + p];
    }
#endif
  };

  v8f o_acc[4] = {};
  float run_max[8], run_sum[8];
#pragma unroll
  for (int r = 0; r < 8; ++r) { run_max[r] = -1e30f; run_sum[r] = 0.f; }

  stage_chunk(0, 0);
#if HAVE_ASYNC_LDS
  __builtin_amdgcn_s_wait_asynccnt(0);
#endif
  __syncthreads();

  for (int kc = 0; kc < kN / 64; ++kc) {
    const int cur = kc & 1;
    if (kc + 1 < kN / 64)                       // prefetch next chunk while computing
      stage_chunk(cur ^ 1, (kc + 1) * 64);

    // S = (Q K^T) * c^-0.5 for this wave's 16 rows x 64 key cols
    v8f s_acc[4];
#pragma unroll
    for (int nt = 0; nt < 4; ++nt) {
      v8f acc = {};
#pragma unroll
      for (int ks = 0; ks < 2; ++ks) {
        const v16h a  = load_a_frag(&qs[wave * 16 + (lane & 15)][ks * 32], lane);
        const v16h bf = load_b_frag(&kTb[cur][nt * 16 + (lane & 15)][ks * 32], lane);
        acc = wmma_f16(a, bf, acc);
      }
      s_acc[nt] = acc;
    }

    // online softmax: row m = r + 8*(lane/16) lives across 16 lanes of a half-wave
#pragma unroll
    for (int r = 0; r < 8; ++r) {
      float mx = run_max[r];
#pragma unroll
      for (int nt = 0; nt < 4; ++nt) {
        s_acc[nt][r] *= 0.125f;                 // 64^-0.5
        mx = fmaxf(mx, s_acc[nt][r]);
      }
      for (int off = 1; off < 16; off <<= 1)
        mx = fmaxf(mx, __shfl_xor(mx, off));
      const float corr = __expf(run_max[r] - mx);
      run_max[r] = mx;
      run_sum[r] *= corr;
#pragma unroll
      for (int ct = 0; ct < 4; ++ct) o_acc[ct][r] *= corr;

      float psum = 0.f;
      const int m = r + ((lane & 16) ? 8 : 0);
#pragma unroll
      for (int nt = 0; nt < 4; ++nt) {
        const float p = __expf(s_acc[nt][r] - mx);
        psum += p;
        ps[wave][m][nt * 16 + (lane & 15)] = (_Float16)p;
      }
      for (int off = 1; off < 16; off <<= 1)
        psum += __shfl_xor(psum, off);
      run_sum[r] += psum;
    }
    __syncthreads();

    // O += P V^T : M = q rows, N = channels (4 tiles), K = 64 key positions
#pragma unroll
    for (int ct = 0; ct < 4; ++ct) {
      v8f acc = o_acc[ct];
#pragma unroll
      for (int kb = 0; kb < 2; ++kb) {
        const v16h a  = load_a_frag(&ps[wave][lane & 15][kb * 32], lane);
        const v16h bf = load_b_frag(&vsb[cur][ct * 16 + (lane & 15)][kb * 32], lane);
        acc = wmma_f16(a, bf, acc);
      }
      o_acc[ct] = acc;
    }

#if HAVE_ASYNC_LDS
    __builtin_amdgcn_s_wait_asynccnt(0);        // next chunk's copies landed
#endif
    __syncthreads();                            // visible to all waves; ps reusable
  }

  // normalize and emit [b, n, c] f16 for the projection GEMM
#pragma unroll
  for (int ct = 0; ct < 4; ++ct) {
    const int ncol = ct * 16 + (lane & 15);
#pragma unroll
    for (int r = 0; r < 8; ++r) {
      const int m = r + ((lane & 16) ? 8 : 0);
      const float o = o_acc[ct][r] / run_sum[r];
      attnT[((size_t)b * kN + q0 + wave * 16 + m) * kC + ncol] = (_Float16)o;
    }
  }
}

// ---------------------------------------------------------------------------
// Kernel 4: projection 1x1 conv + bias + residual, fp32 output.
// ---------------------------------------------------------------------------
__global__ __launch_bounds__(128) void proj_kernel(
    const float* __restrict__ x, const _Float16* __restrict__ attnT,
    const float* __restrict__ proj_w, const float* __restrict__ proj_b,
    float* __restrict__ out) {
  const int blk = blockIdx.x;
  const int b   = blk >> 6;
  const int p0  = (blk & 63) * 64;
  __shared__ _Float16 at[64][LDSP];        // [pos][chan]
  __shared__ _Float16 wl[64][LDSP];        // [out][in]
  const int tid = threadIdx.x, lane = tid & 31, wave = tid >> 5;

  for (int i = tid; i < 64 * 64; i += 128) {
    const int p = i >> 6, c = i & 63;
    at[p][c] = attnT[((size_t)b * kN + p0 + p) * kC + c];
    wl[p][c] = (_Float16)proj_w[i];
  }
  __syncthreads();

#pragma unroll
  for (int nt = 0; nt < 4; ++nt) {
    v8f acc = {};
#pragma unroll
    for (int ks = 0; ks < 2; ++ks) {
      const v16h a  = load_a_frag(&at[wave * 16 + (lane & 15)][ks * 32], lane);
      const v16h bf = load_b_frag(&wl[nt * 16 + (lane & 15)][ks * 32], lane);
      acc = wmma_f16(a, bf, acc);
    }
    const int o = nt * 16 + (lane & 15);
    const float bias = proj_b[o];
#pragma unroll
    for (int r = 0; r < 8; ++r) {
      const int m   = r + ((lane & 16) ? 8 : 0);
      const int pos = p0 + wave * 16 + m;
      const size_t idx = ((size_t)b * kC + o) * kN + pos;
      out[idx] = x[idx] + acc[r] + bias;
    }
  }
}

// ---------------------------------------------------------------------------
extern "C" void kernel_launch(void* const* d_in, const int* in_sizes, int n_in,
                              void* d_out, int out_size, void* d_ws, size_t ws_size,
                              hipStream_t stream) {
  const float* x      = (const float*)d_in[0];
  const float* norm_w = (const float*)d_in[1];
  const float* norm_b = (const float*)d_in[2];
  const float* q_w    = (const float*)d_in[3];
  const float* q_b    = (const float*)d_in[4];
  const float* k_w    = (const float*)d_in[5];
  const float* k_b    = (const float*)d_in[6];
  const float* v_w    = (const float*)d_in[7];
  const float* v_b    = (const float*)d_in[8];
  const float* proj_w = (const float*)d_in[9];
  const float* proj_b = (const float*)d_in[10];

  const size_t NC = (size_t)kB * kN * kC;            // 524288 elements
  float*    stats = (float*)d_ws;                    // 64 floats
  _Float16* qT    = (_Float16*)((char*)d_ws + 1024);
  _Float16* kTg   = qT + NC;
  _Float16* vG    = kTg + NC;
  _Float16* attnT = vG + NC;
  (void)in_sizes; (void)n_in; (void)out_size; (void)ws_size;

  gn_stats_kernel<<<kB * kNG, 256, 0, stream>>>(x, stats);
  qkv_kernel<<<kB * (kN / 64), 128, 0, stream>>>(x, stats, norm_w, norm_b,
                                                 q_w, q_b, k_w, k_b, v_w, v_b,
                                                 qT, kTg, vG);
  attn_kernel<<<kB * (kN / 64), 128, 0, stream>>>(qT, kTg, vG, attnT);
  proj_kernel<<<kB * (kN / 64), 128, 0, stream>>>(x, attnT, proj_w, proj_b,
                                                  (float*)d_out);
}
